// DyG4SR_25649544691796
// MI455X (gfx1250) — compile-verified, matrix operand order checked by reading, not run on hardware
//
// DyG4SR for MI455X (gfx1250): gather -> bf16 pack -> WMMA GEMM pipeline.
//
// Roofline reasoning (MI455X): total dense GEMM work across the 5 branches is
// ~6.8e11 flops -- negligible against gfx1250 WMMA bf16 throughput. The
// pipeline is bound by data movement: embedding tables (user 51MB / item 26MB)
// live in the 192MB L2, and ~0.9GB of intermediates stream through HBM at
// 23.3 TB/s (~40us floor). Hence: pack activations as bf16 (halves traffic,
// native V_WMMA_F32_16X16X32_BF16 operand type), accumulate in f32, and keep
// softmax as a tiny wave32 shuffle/LDS kernel.

#include <hip/hip_runtime.h>
#include <hip/hip_bf16.h>

#define DEV __device__ __forceinline__

typedef __attribute__((ext_vector_type(16))) __bf16 v16bf;
typedef __attribute__((ext_vector_type(8)))  float  v8f;

namespace {
constexpr int kB     = 2048;
constexpr int kD     = 128;
constexpr int kE     = 160000;
constexpr int kShots = 4;
constexpr int kEShot = kE / kShots;   // 40000
constexpr int kNeig  = 50;
constexpr int kTB    = 20;
constexpr int kM1    = kB * 20;       // 40960 inner source rows
constexpr int kM1N   = kM1 * 10;      // 409600 inner neighbor rows
}

// ---------- bf16 helpers (RNE, bit-level; no __bf16 scalar arithmetic) ------
DEV unsigned short f2bf_bits(float x) {
  unsigned u = __float_as_uint(x);
  unsigned r = 0x7FFFu + ((u >> 16) & 1u);
  return (unsigned short)((u + r) >> 16);
}
DEV float bf2f(unsigned short b) {
  return __uint_as_float(((unsigned)b) << 16);
}
DEV int time_bucket(int d) {
  float a = fabsf((float)d) + 1.0f;
  int b = (int)__log2f(a);            // a>=1 so trunc == floor, matches astype(int32)
  return min(max(b, 0), kTB - 1);
}

// ---------- WMMA GEMM: C[M,NOUT] = A[M,KD](bf16) @ Wt^T (Wt is [NOUT][KD] bf16)
struct Pack32 { uint4 a, b; };

DEV v16bf load_frag_bf16(const unsigned short* rowptr, int koff) {
  Pack32 p;
  p.a = *reinterpret_cast<const uint4*>(rowptr + koff);        // k = koff..koff+7
  p.b = *reinterpret_cast<const uint4*>(rowptr + koff + 16);   // k = koff+16..+23
  return __builtin_bit_cast(v16bf, p);
}

template<int KD, int NOUT, bool BIAS, bool RELU, bool OBF16>
__global__ __launch_bounds__(256)
void gemm_bf16_wmma(const unsigned short* __restrict__ A,
                    const unsigned short* __restrict__ Wt,
                    const float* __restrict__ bias,
                    void* __restrict__ Cout)
{
  constexpr int WN = NOUT / 64;   // waves along N (64-wide wave tile)
  constexpr int WM = 8 / WN;      // waves along M
  const int tid  = threadIdx.x;
  const int wave = tid >> 5;
  const int lane = tid & 31;
  const int hi   = lane >> 4;     // half-wave select (k-group)
  const int ln   = lane & 15;
  const int wn   = wave % WN;
  const int wm   = wave / WN;
  const int m0   = blockIdx.x * (WM * 16) + wm * 16;
  const int n0   = wn * 64;

  const unsigned short* arow = A  + (size_t)(m0 + ln) * KD;
  const unsigned short* b0r  = Wt + (size_t)(n0 +  0 + ln) * KD;
  const unsigned short* b1r  = Wt + (size_t)(n0 + 16 + ln) * KD;
  const unsigned short* b2r  = Wt + (size_t)(n0 + 32 + ln) * KD;
  const unsigned short* b3r  = Wt + (size_t)(n0 + 48 + ln) * KD;

  v8f acc0 = {}, acc1 = {}, acc2 = {}, acc3 = {};
#pragma unroll
  for (int k0 = 0; k0 < KD; k0 += 32) {
    const int koff = k0 + 8 * hi;
    if (k0 + 32 < KD) __builtin_prefetch(arow + koff + 32, 0, 0);
    v16bf af  = load_frag_bf16(arow, koff);
    v16bf bf0 = load_frag_bf16(b0r, koff);
    v16bf bf1 = load_frag_bf16(b1r, koff);
    v16bf bf2 = load_frag_bf16(b2r, koff);
    v16bf bf3 = load_frag_bf16(b3r, koff);
    acc0 = __builtin_amdgcn_wmma_f32_16x16x32_bf16(false, af, false, bf0, (short)0, acc0, false, false);
    acc1 = __builtin_amdgcn_wmma_f32_16x16x32_bf16(false, af, false, bf1, (short)0, acc1, false, false);
    acc2 = __builtin_amdgcn_wmma_f32_16x16x32_bf16(false, af, false, bf2, (short)0, acc2, false, false);
    acc3 = __builtin_amdgcn_wmma_f32_16x16x32_bf16(false, af, false, bf3, (short)0, acc3, false, false);
  }

  v8f accs[4] = {acc0, acc1, acc2, acc3};
#pragma unroll
  for (int t = 0; t < 4; ++t) {
#pragma unroll
    for (int g = 0; g < 8; ++g) {
      const int r = m0 + g + 8 * hi;        // C layout: VGPR g -> M=g (lanes 0-15) / g+8
      const int c = n0 + t * 16 + ln;
      float v = accs[t][g];
      if constexpr (BIAS) v += bias[c];
      if constexpr (RELU) v = fmaxf(v, 0.0f);
      if constexpr (OBF16) ((unsigned short*)Cout)[(size_t)r * NOUT + c] = f2bf_bits(v);
      else                 ((float*)Cout)[(size_t)r * NOUT + c] = v;
    }
  }
}

// ---------- weight prep: f32 [R,C] -> bf16 transposed [C,R] ------------------
__global__ void transpose_bf16_kernel(const float* __restrict__ src,
                                      unsigned short* __restrict__ dst,
                                      int R, int C)
{
  int idx = blockIdx.x * 256 + threadIdx.x;
  if (idx >= R * C) return;
  int r = idx / C, c = idx % C;
  dst[(size_t)c * R + r] = f2bf_bits(src[idx]);
}

// ---------- level-1 input builder (one block per inner source row) ----------
__global__ __launch_bounds__(256)
void build_l1_kernel(const int* __restrict__ edges, const int* __restrict__ ts,
                     const int* __restrict__ t_neig, const int* __restrict__ t_edge,
                     const int* __restrict__ t_time, const float* __restrict__ t_mask,
                     const int* __restrict__ m_neig, const int* __restrict__ m_edge,
                     const int* __restrict__ m_time, const float* __restrict__ m_mask,
                     const float* __restrict__ item_emb, const float* __restrict__ user_emb,
                     const float* __restrict__ time_emb,
                     unsigned short* __restrict__ q_in1, unsigned short* __restrict__ k_in1,
                     float* __restrict__ kmask1, int* __restrict__ top_bucket,
                     float* __restrict__ top_mask, int shot)
{
  const int r = blockIdx.x;           // 0..kM1-1
  const int i = r / 20, j = r % 20;   // batch row / top-neighbor column
  const int t = threadIdx.x;

  int e  = edges[i];
  int lo = 0;
  if (shot >= 0) { lo = shot * kEShot; int el = e - lo; e = min(max(el, 0), kEShot - 1); }
  const size_t base = (size_t)e * kNeig + (kNeig - 20 + j);   // cols 30..49

  int   u_item = t_neig[base];
  int   e1     = t_edge[base] + (shot >= 0 ? lo : 0);         // global edge id
  int   t1     = t_time[base];
  float m1     = t_mask[base];

  if (t == 0) {
    top_bucket[r] = time_bucket(ts[i] - t1);
    top_mask[r]   = m1;
  }
  // q_in1 = [item_emb[u_item] | time_emb[0]]  (bf16)
  float qv = (t < kD) ? item_emb[(size_t)u_item * kD + t] : time_emb[t - kD];
  q_in1[(size_t)r * 256 + t] = f2bf_bits(qv);

  // mid (item-side) table row; for ddyg this undoes the +lo (always in range)
  int e2 = e1;
  if (shot >= 0) { int el2 = e1 - lo; e2 = min(max(el2, 0), kEShot - 1); }
  const size_t mb = (size_t)e2 * kNeig;
#pragma unroll
  for (int c = 0; c < 10; ++c) {
    const size_t o  = mb + (kNeig - 10 + c);                  // cols 40..49
    int   vu = m_neig[o];
    int   t2 = m_time[o];
    float m2 = m_mask[o];
    int   b2 = time_bucket(t1 - t2);
    const size_t rr = (size_t)r * 10 + c;
    if (t == 0) kmask1[rr] = m2;
    float kv = (t < kD) ? user_emb[(size_t)vu * kD + t]
                        : time_emb[(size_t)b2 * kD + (t - kD)];
    k_in1[rr * 256 + t] = f2bf_bits(kv);
  }
}

// ---------- attention core: scores + softmax + weighted V (one block/row) ---
template<int N>
__global__ __launch_bounds__(256)
void attn_kernel(const float* __restrict__ q, const unsigned short* __restrict__ k,
                 const unsigned short* __restrict__ v, const float* __restrict__ kmask,
                 unsigned short* __restrict__ o_att, int* __restrict__ allpad)
{
  const int r    = blockIdx.x;
  const int tid  = threadIdx.x;
  const int wave = tid >> 5;
  const int lane = tid & 31;
  __shared__ float s_sc[2][32];
  __shared__ float s_w[2][32];
  const float scale = 0.08838834764831845f;   // 1/sqrt(128)

  // scores[h][n] = q_h . k_{n,h}  (wave per (n,h) pair, shuffle-reduced)
  for (int p = wave; p < 2 * N; p += 8) {
    const int h = p & 1, n = p >> 1;
    const float*          qp = q + (size_t)r * 256 + h * kD;
    const unsigned short* kp = k + ((size_t)r * N + n) * 256 + h * kD;
    float acc = 0.0f;
    for (int dd = lane; dd < kD; dd += 32) acc += qp[dd] * bf2f(kp[dd]);
    for (int off = 16; off > 0; off >>= 1) acc += __shfl_xor(acc, off, 32);
    if (lane == 0) {
      float msk = kmask[(size_t)r * N + n];
      s_sc[h][n] = (msk > 0.0f) ? -1e9f : acc * scale;
    }
  }
  if (tid == 0) {
    int ap = 1;
    for (int n = 0; n < N; ++n) if (!(kmask[(size_t)r * N + n] > 0.0f)) { ap = 0; break; }
    allpad[r] = ap;
  }
  __syncthreads();

  // softmax over N, one wave per head
  if (wave < 2) {
    const int h = wave;
    float s = (lane < N) ? s_sc[h][lane] : -3.0e38f;
    float m = s;
    for (int off = 16; off > 0; off >>= 1) m = fmaxf(m, __shfl_xor(m, off, 32));
    float e = (lane < N) ? __expf(s - m) : 0.0f;
    float sum = e;
    for (int off = 16; off > 0; off >>= 1) sum += __shfl_xor(sum, off, 32);
    if (lane < N) s_w[h][lane] = e / sum;
  }
  __syncthreads();

  // o[j] = sum_n w[h][n] * v[n][j]
  const int j = tid, h = j >> 7;
  float acc = 0.0f;
#pragma unroll
  for (int n = 0; n < N; ++n)
    acc += s_w[h][n] * bf2f(v[((size_t)r * N + n) * 256 + j]);
  o_att[(size_t)r * 256 + j] = f2bf_bits(acc);
}

// ---------- h_in = [allpad ? 0 : o@Wo | src_feat] (bf16, [M,384]) -----------
__global__ void build_hin_kernel(const float* __restrict__ o2,
                                 const unsigned short* __restrict__ q_in,
                                 const int* __restrict__ allpad,
                                 unsigned short* __restrict__ h_in, int M)
{
  int idx = blockIdx.x * 256 + threadIdx.x;
  if (idx >= M * 384) return;
  int r = idx / 384, j = idx % 384;
  unsigned short out;
  if (j < 256) out = allpad[r] ? (unsigned short)0 : f2bf_bits(o2[(size_t)r * 256 + j]);
  else         out = q_in[(size_t)r * 256 + (j - 256)];   // bf16 src_feat copy
  h_in[idx] = out;
}

// ---------- level-2 input builder (one block per batch row) -----------------
__global__ __launch_bounds__(256)
void build_l2_kernel(const int* __restrict__ nodes, const float* __restrict__ user_emb,
                     const float* __restrict__ time_emb, const float* __restrict__ n1,
                     const int* __restrict__ top_bucket,
                     unsigned short* __restrict__ q_in2, unsigned short* __restrict__ k_in2)
{
  const int i = blockIdx.x, t = threadIdx.x;
  float qv = (t < kD) ? user_emb[(size_t)nodes[i] * kD + t] : time_emb[t - kD];
  q_in2[(size_t)i * 256 + t] = f2bf_bits(qv);
#pragma unroll 4
  for (int j = 0; j < 20; ++j) {
    const size_t rr = (size_t)i * 20 + j;
    float kv = (t < kD) ? n1[rr * kD + t]
                        : time_emb[(size_t)top_bucket[rr] * kD + (t - kD)];
    k_in2[rr * 256 + t] = f2bf_bits(kv);
  }
}

// ---------- finalize: ddyg valid-mask and write output slice ----------------
__global__ void finalize_kernel(const float* __restrict__ out2, const int* __restrict__ edges,
                                const int* __restrict__ nodes, const float* __restrict__ user_emb,
                                float* __restrict__ dst, int shot)
{
  int idx = blockIdx.x * 256 + threadIdx.x;
  if (idx >= kB * kD) return;
  int i = idx >> 7, d = idx & (kD - 1);
  float v = out2[idx];
  if (shot >= 0) {
    int e = edges[i], lo = shot * kEShot;
    bool valid = (e >= lo) && (e <= lo + kEShot - 1);
    if (!valid) v = user_emb[(size_t)nodes[i] * kD + d];
  }
  dst[idx] = v;
}

// ---------- host-side GEMM launcher -----------------------------------------
template<int KD, int NOUT, bool BIAS, bool RELU, bool OBF16>
static void launch_gemm(const void* A, const void* Wt, const float* bias, void* C,
                        int M, hipStream_t s)
{
  constexpr int WN = NOUT / 64;
  constexpr int WM = 8 / WN;
  dim3 grid(M / (WM * 16));
  gemm_bf16_wmma<KD, NOUT, BIAS, RELU, OBF16><<<grid, 256, 0, s>>>(
      (const unsigned short*)A, (const unsigned short*)Wt, bias, C);
}

extern "C" void kernel_launch(void* const* d_in, const int* in_sizes, int n_in,
                              void* d_out, int out_size, void* d_ws, size_t ws_size,
                              hipStream_t stream)
{
  (void)in_sizes; (void)n_in; (void)out_size; (void)ws_size;

  // Input order = setup_inputs() insertion order, leaves flattened in place.
  const int*   nodes    = (const int*)  d_in[0];
  const int*   edges    = (const int*)  d_in[1];
  const int*   ts       = (const int*)  d_in[2];
  const float* user_emb = (const float*)d_in[4];
  const float* item_emb = (const float*)d_in[5];
  const float* time_emb = (const float*)d_in[6];
  const float* Wq = (const float*)d_in[7];
  const float* Wk = (const float*)d_in[8];
  const float* Wv = (const float*)d_in[9];
  const float* Wo = (const float*)d_in[10];
  const float* W1 = (const float*)d_in[11];
  const float* b1 = (const float*)d_in[12];
  const float* W2 = (const float*)d_in[13];
  const float* b2 = (const float*)d_in[14];
  // tables: {neig, edge, time, mask} each
  const int idxUT = 15, idxIT = 19, idxDU = 23, idxDI = 39;

  // ---- deterministic workspace arena ----
  size_t off = 0;
  auto alloc = [&](size_t bytes) -> void* {
    void* p = (char*)d_ws + off;
    off += (bytes + 255) & ~(size_t)255;
    return p;
  };
  unsigned short* wq_t = (unsigned short*)alloc(2 * 256 * 256 * 2);
  unsigned short* wk_t = (unsigned short*)alloc(2 * 256 * 256 * 2);
  unsigned short* wv_t = (unsigned short*)alloc(2 * 256 * 256 * 2);
  unsigned short* wo_t = (unsigned short*)alloc(2 * 256 * 256 * 2);
  unsigned short* w1_t = (unsigned short*)alloc(2 * 384 * 128 * 2);
  unsigned short* w2_t = (unsigned short*)alloc(2 * 128 * 128 * 2);

  unsigned short* q_in1  = (unsigned short*)alloc((size_t)kM1  * 256 * 2);
  unsigned short* k_in1  = (unsigned short*)alloc((size_t)kM1N * 256 * 2);
  float*          kmask1 = (float*)         alloc((size_t)kM1N * 4);
  int*            top_bk = (int*)           alloc((size_t)kM1 * 4);
  float*          top_mk = (float*)         alloc((size_t)kM1 * 4);
  int*            allp1  = (int*)           alloc((size_t)kM1 * 4);
  float*          q1     = (float*)         alloc((size_t)kM1  * 256 * 4);
  unsigned short* k1     = (unsigned short*)alloc((size_t)kM1N * 256 * 2);
  unsigned short* v1     = (unsigned short*)alloc((size_t)kM1N * 256 * 2);
  unsigned short* oatt1  = (unsigned short*)alloc((size_t)kM1 * 256 * 2);
  float*          o2_1   = (float*)         alloc((size_t)kM1 * 256 * 4);
  unsigned short* hin1   = (unsigned short*)alloc((size_t)kM1 * 384 * 2);
  unsigned short* h1     = (unsigned short*)alloc((size_t)kM1 * 128 * 2);
  float*          n1     = (float*)         alloc((size_t)kM1 * 128 * 4);

  unsigned short* q_in2  = (unsigned short*)alloc((size_t)kB  * 256 * 2);
  unsigned short* k_in2  = (unsigned short*)alloc((size_t)kM1 * 256 * 2);
  float*          q2     = (float*)         alloc((size_t)kB  * 256 * 4);
  unsigned short* k2     = (unsigned short*)alloc((size_t)kM1 * 256 * 2);
  unsigned short* v2     = (unsigned short*)alloc((size_t)kM1 * 256 * 2);
  unsigned short* oatt2  = (unsigned short*)alloc((size_t)kB * 256 * 2);
  int*            allp2  = (int*)           alloc((size_t)kB * 4);
  float*          o2_2   = (float*)         alloc((size_t)kB * 256 * 4);
  unsigned short* hin2   = (unsigned short*)alloc((size_t)kB * 384 * 2);
  unsigned short* h2     = (unsigned short*)alloc((size_t)kB * 128 * 2);
  float*          out2   = (float*)         alloc((size_t)kB * 128 * 4);

  // ---- weight prep: transpose to bf16 [Nout][K] (contiguous B fragments) ----
  for (int l = 0; l < 2; ++l) {
    int g = (256 * 256 + 255) / 256;
    transpose_bf16_kernel<<<g, 256, 0, stream>>>(Wq + (size_t)l * 65536, wq_t + (size_t)l * 65536, 256, 256);
    transpose_bf16_kernel<<<g, 256, 0, stream>>>(Wk + (size_t)l * 65536, wk_t + (size_t)l * 65536, 256, 256);
    transpose_bf16_kernel<<<g, 256, 0, stream>>>(Wv + (size_t)l * 65536, wv_t + (size_t)l * 65536, 256, 256);
    transpose_bf16_kernel<<<g, 256, 0, stream>>>(Wo + (size_t)l * 65536, wo_t + (size_t)l * 65536, 256, 256);
    transpose_bf16_kernel<<<(384 * 128 + 255) / 256, 256, 0, stream>>>(W1 + (size_t)l * 49152, w1_t + (size_t)l * 49152, 384, 128);
    transpose_bf16_kernel<<<(128 * 128 + 255) / 256, 256, 0, stream>>>(W2 + (size_t)l * 16384, w2_t + (size_t)l * 16384, 128, 128);
  }

  // ---- branches: 0 = main, 1..4 = ddyg shots (sequential, arena reuse) ----
  for (int br = 0; br < 5; ++br) {
    const int shot = br - 1;
    const int base_u = (shot < 0) ? idxUT : idxDU + 4 * shot;   // top (user-side) table
    const int base_i = (shot < 0) ? idxIT : idxDI + 4 * shot;   // mid (item-side) table
    const int*   tn = (const int*)  d_in[base_u + 0];
    const int*   te = (const int*)  d_in[base_u + 1];
    const int*   tt = (const int*)  d_in[base_u + 2];
    const float* tm = (const float*)d_in[base_u + 3];
    const int*   mn = (const int*)  d_in[base_i + 0];
    const int*   me = (const int*)  d_in[base_i + 1];
    const int*   mt = (const int*)  d_in[base_i + 2];
    const float* mm = (const float*)d_in[base_i + 3];

    // ---- level 1 (l=0 weights): 40960 rows x 10 neighbors ----
    build_l1_kernel<<<kM1, 256, 0, stream>>>(edges, ts, tn, te, tt, tm, mn, me, mt, mm,
                                             item_emb, user_emb, time_emb,
                                             q_in1, k_in1, kmask1, top_bk, top_mk, shot);
    launch_gemm<256, 256, false, false, false>(q_in1, wq_t, nullptr, q1, kM1,  stream);
    launch_gemm<256, 256, false, false, true >(k_in1, wk_t, nullptr, k1, kM1N, stream);
    launch_gemm<256, 256, false, false, true >(k_in1, wv_t, nullptr, v1, kM1N, stream);
    attn_kernel<10><<<kM1, 256, 0, stream>>>(q1, k1, v1, kmask1, oatt1, allp1);
    launch_gemm<256, 256, false, false, false>(oatt1, wo_t, nullptr, o2_1, kM1, stream);
    build_hin_kernel<<<(kM1 * 384 + 255) / 256, 256, 0, stream>>>(o2_1, q_in1, allp1, hin1, kM1);
    launch_gemm<384, 128, true, true,  true >(hin1, w1_t, b1, h1, kM1, stream);
    launch_gemm<128, 128, true, false, false>(h1,   w2_t, b2, n1, kM1, stream);
    // (ddyg inner valid-mask is provably all-true: shot-local edge ids + lo
    //  always land inside [lo, hi], so out==attn output at this level.)

    // ---- level 2 (l=1 weights): 2048 rows x 20 neighbors ----
    build_l2_kernel<<<kB, 256, 0, stream>>>(nodes, user_emb, time_emb, n1, top_bk, q_in2, k_in2);
    launch_gemm<256, 256, false, false, false>(q_in2, wq_t + 65536, nullptr, q2, kB,  stream);
    launch_gemm<256, 256, false, false, true >(k_in2, wk_t + 65536, nullptr, k2, kM1, stream);
    launch_gemm<256, 256, false, false, true >(k_in2, wv_t + 65536, nullptr, v2, kM1, stream);
    attn_kernel<20><<<kB, 256, 0, stream>>>(q2, k2, v2, top_mk, oatt2, allp2);
    launch_gemm<256, 256, false, false, false>(oatt2, wo_t + 65536, nullptr, o2_2, kB, stream);
    build_hin_kernel<<<(kB * 384 + 255) / 256, 256, 0, stream>>>(o2_2, q_in2, allp2, hin2, kB);
    launch_gemm<384, 128, true, true,  true >(hin2, w1_t + 49152, b1 + 128, h2,   kB, stream);
    launch_gemm<128, 128, true, false, false>(h2,   w2_t + 16384, b2 + 128, out2, kB, stream);

    float* dst = (float*)d_out + (size_t)br * kB * kD;
    finalize_kernel<<<(kB * kD + 255) / 256, 256, 0, stream>>>(out2, edges, nodes, user_emb, dst, shot);
  }
}
